// DecoderAttentionParallel_41918880808980
// MI455X (gfx1250) — matrix operand order repaired and checked by reading, not compiled, not run on hardware
//
#include <hip/hip_runtime.h>

// Problem constants (B, NC, C, DC, H, W = 2, 4, 64, 128, 128, 128)
#define Bc   2
#define NCc  4
#define Cc   64
#define DCc  128
#define Hc   128
#define Wc   128
#define HWc  (Hc * Wc)   // 16384

typedef __attribute__((ext_vector_type(16))) __bf16 bf16x16;
typedef __attribute__((ext_vector_type(8)))  float  f32x8;
typedef unsigned int uint32x4 __attribute__((ext_vector_type(4)));
typedef int          int32x4  __attribute__((ext_vector_type(4)));
typedef int          int32x8  __attribute__((ext_vector_type(8)));

// ---- CDNA5 Tensor Data Mover availability -------------------------------
#if defined(__has_builtin)
#  if __has_builtin(__builtin_amdgcn_tensor_load_to_lds) && \
      __has_builtin(__builtin_amdgcn_s_wait_tensorcnt)
#    define HAVE_TDM 1
#  endif
#endif
#ifndef HAVE_TDM
#  define HAVE_TDM 0
#endif

// LDS pitch for the 32xK x 128px staging tile.
#if HAVE_TDM
#  define PITCH_X 130   // TDM pad: +2 DWORDs after each 128-DW row
#else
#  define PITCH_X 128   // float4-aligned fallback staging
#endif
#define PITCH_V 131     // conv tile (1 halo col each side + 1 pad)

__device__ __forceinline__ unsigned lds_addr_of(const void* p) {
  return (unsigned)(size_t)p;   // low 32 bits of generic LDS pointer = LDS offset
}

#if HAVE_TDM
// Issue a 2-D TDM tile load: tile0 x tile1 elements of 4B, global row stride
// (elements) stride0, LDS row padding via padbits.
__device__ __forceinline__ void tdm_stage(unsigned lds_byte, const float* g,
                                          unsigned tile0, unsigned tile1,
                                          unsigned long long stride0,
                                          unsigned padbits) {
  const unsigned long long ga = (unsigned long long)(size_t)g;
  uint32x4 g0;
  g0[0] = 1u;                                   // count = 1 valid descriptor
  g0[1] = lds_byte;                             // lds_addr
  g0[2] = (unsigned)ga;                         // global_addr[31:0]
  g0[3] = ((unsigned)(ga >> 32) & 0x01FFFFFFu)  // global_addr[56:32]
          | 0x80000000u;                        // type = 2 ("image")
  const unsigned td0 = 1u << 30, td1 = 1u << 20;  // huge dims: no OOB zeroing
  int32x8 g1;
  g1[0] = (int)(0x00020000u | padbits);         // data_size=4B (+pad cfg)
  g1[1] = (int)((td0 & 0xFFFFu) << 16);         // tensor_dim0 lo
  g1[2] = (int)((td0 >> 16) | ((td1 & 0xFFFFu) << 16));
  g1[3] = (int)((td1 >> 16) | (tile0 << 16));   // tile_dim0
  g1[4] = (int)(tile1 & 0xFFFFu);               // tile_dim1 (tile_dim2 = 0)
  g1[5] = (int)(unsigned)stride0;               // tensor_dim0_stride lo32
  g1[6] = (int)((unsigned)(stride0 >> 32) & 0xFFFFu);
  g1[7] = 0;
  int32x4 gz = {0, 0, 0, 0};
#  if defined(__clang_major__) && __clang_major__ >= 23
  int32x8 gz8 = {0, 0, 0, 0, 0, 0, 0, 0};
  __builtin_amdgcn_tensor_load_to_lds(g0, g1, gz, gz, gz8, 0);
#  else
  __builtin_amdgcn_tensor_load_to_lds(g0, g1, gz, gz, 0);
#  endif
}
#endif

// Issue staging of a [32 k] x [128 px] f32 tile (global row stride HWc).
__device__ __forceinline__ void stage_issue(float* sbuf, const float* gsrc, int tid) {
#if HAVE_TDM
  if (tid < 32)  // pad_enable | interval=128 DW (6) | amount=2 DW (1) -> pitch 130
    tdm_stage(lds_addr_of(sbuf), gsrc, 128u, 32u, (unsigned long long)HWc,
              (1u << 20) | (6u << 22) | (1u << 25));
#else
#pragma unroll
  for (int pass = 0; pass < 4; ++pass) {
    const int r  = (tid >> 5) + pass * 8;
    const int cv = (tid & 31) * 4;
    const float4 v = *(const float4*)(gsrc + r * HWc + cv);
    *(float4*)&sbuf[r * PITCH_X + cv] = v;
  }
#endif
}

// Arrival barrier: TDM data landed + all waves synchronized.
__device__ __forceinline__ void stage_wait(int tid) {
#if HAVE_TDM
  if (tid < 32) __builtin_amdgcn_s_wait_tensorcnt(0);
#endif
  __syncthreads();
}

// ---------------------------------------------------------------------------
// WMMA helpers (wave32, v_wmma_f32_16x16x32_bf16)
// ---------------------------------------------------------------------------
__device__ __forceinline__ f32x8 wmma_bf16(bf16x16 a, bf16x16 b, f32x8 c) {
  return __builtin_amdgcn_wmma_f32_16x16x32_bf16(false, a, false, b, (short)0, c,
                                                 false, false);
}

// Pre-packed A fragment: one 32B vector load per (chunk, mb).
__device__ __forceinline__ bf16x16 load_a_packed(const __bf16* __restrict__ Ap,
                                                 int chunk, int mb, int lane) {
  return *(const bf16x16*)(Ap + (size_t)(((chunk * 4 + mb) * 32 + lane) << 4));
}

// B fragment (32x16 bf16) from the LDS f32 tile.
// lanes 0-15 -> col=lane, K=0..7,16..23 ; lanes 16-31 -> col=lane-16, K=8..15,24..31.
__device__ __forceinline__ bf16x16 b_frag_lds(const float* __restrict__ s, int pitch,
                                              int pcol, int lane) {
  const int kb = (lane >> 4) * 8;
  float t[16];
#pragma unroll
  for (int i = 0; i < 16; ++i) t[i] = s[(kb + (i & 7) + (i >> 3) * 16) * pitch + pcol];
  bf16x16 b;
#pragma unroll
  for (int i = 0; i < 16; ++i) b[i] = (__bf16)t[i];
  return b;
}

// C/D 16x16 f32 store: lane -> col=lane&15 ; VGPR r -> row r + (lane>=16 ? 8:0).
__device__ __forceinline__ void store_tile(float* __restrict__ Out, int p0,
                                           const f32x8* acc,
                                           const float* __restrict__ bias,
                                           int lane, bool leaky) {
  const int colp = p0 + (lane & 15);
  const int radd = (lane >> 4) * 8;
#pragma unroll
  for (int mb = 0; mb < 4; ++mb) {
#pragma unroll
    for (int r = 0; r < 8; ++r) {
      const int row = mb * 16 + radd + r;
      float v = acc[mb][r] + bias[row];
      if (leaky) v = (v >= 0.f) ? v : 0.2f * v;
      Out[row * HWc + colp] = v;
    }
  }
}

// ---------------------------------------------------------------------------
// Weight pre-packing into WMMA A-fragment layout (bf16).
// Element e: j=e&1, r=(e>>1)&7, lane=(e>>4)&31, mb=(e>>9)&3, chunk=e>>11.
// m = mb*16 + (lane&15); kk = chunk*32 + (lane>>4)*8 + (r>>2)*16 + (r&3)*2 + j.
// ---------------------------------------------------------------------------
__global__ __launch_bounds__(256) void k_pack_plain(const float* __restrict__ src,
                                                    __bf16* __restrict__ dst, int ld) {
  const int e    = blockIdx.x * 256 + threadIdx.x;
  const int j    = e & 1;
  const int r    = (e >> 1) & 7;
  const int lane = (e >> 4) & 31;
  const int mb   = (e >> 9) & 3;
  const int ch   = e >> 11;
  const int m  = mb * 16 + (lane & 15);
  const int kk = ch * 32 + ((lane >> 4) * 8) + ((r >> 2) * 16) + ((r & 3) * 2) + j;
  dst[e] = (__bf16)src[m * ld + kk];
}

// W_val (C, DC, 3, 3): chunk id cc = tap*4 + c ; src = (m*DC + kk)*9 + tap.
__global__ __launch_bounds__(256) void k_pack_wval(const float* __restrict__ src,
                                                   __bf16* __restrict__ dst) {
  const int e    = blockIdx.x * 256 + threadIdx.x;
  const int j    = e & 1;
  const int r    = (e >> 1) & 7;
  const int lane = (e >> 4) & 31;
  const int mb   = (e >> 9) & 3;
  const int cc   = e >> 11;
  const int tap  = cc >> 2;
  const int c    = cc & 3;
  const int m  = mb * 16 + (lane & 15);
  const int kk = c * 32 + ((lane >> 4) * 8) + ((r >> 2) * 16) + ((r & 3) * 2) + j;
  dst[e] = (__bf16)src[(m * DCc + kk) * 9 + tap];
}

// ---------------------------------------------------------------------------
// Kernel 1: queries[b,c,p] = sum_d W_dec[c,d]*dec[b,d,p] + b_dec[c]
// Double-buffered TDM staging.
// ---------------------------------------------------------------------------
__global__ __launch_bounds__(256, 1) void k_queries(const float* __restrict__ dec,
                                                    const __bf16* __restrict__ Apack,
                                                    const float* __restrict__ bdec,
                                                    float* __restrict__ qbuf) {
  __shared__ float sbuf[2][32 * PITCH_X];
  const int tid = threadIdx.x, lane = tid & 31, wave = tid >> 5;
  const int b    = blockIdx.y;
  const int pblk = blockIdx.x * 128;
  const int pcol = wave * 16 + (lane & 15);
  const float* X = dec + (size_t)b * DCc * HWc + pblk;
  float* Out     = qbuf + (size_t)b * Cc * HWc;

  f32x8 acc[4] = {};
  stage_issue(sbuf[0], X, tid);
#pragma unroll 1
  for (int ch = 0; ch < 4; ++ch) {            // K = 128
    stage_wait(tid);
    if (ch + 1 < 4) stage_issue(sbuf[(ch + 1) & 1], X + (ch + 1) * 32 * HWc, tid);
    bf16x16 bf = b_frag_lds(sbuf[ch & 1], PITCH_X, pcol, lane);
#pragma unroll
    for (int mb = 0; mb < 4; ++mb)
      acc[mb] = wmma_bf16(load_a_packed(Apack, ch, mb, lane), bf, acc[mb]);
  }
  store_tile(Out, pblk + wave * 16, acc, bdec, lane, false);
}

// ---------------------------------------------------------------------------
// Kernel 2: keys[b,n,e,p] = sum_c W_enc[e,c]*ctx[b,n,c,p] + b_enc[e]
// ---------------------------------------------------------------------------
__global__ __launch_bounds__(256, 1) void k_keys(const float* __restrict__ ctxs,
                                                 const __bf16* __restrict__ Apack,
                                                 const float* __restrict__ benc,
                                                 float* __restrict__ kbuf) {
  __shared__ float sbuf[2][32 * PITCH_X];
  const int tid = threadIdx.x, lane = tid & 31, wave = tid >> 5;
  const int bn   = blockIdx.y;
  const int pblk = blockIdx.x * 128;
  const int pcol = wave * 16 + (lane & 15);
  const float* X = ctxs + (size_t)bn * Cc * HWc + pblk;
  float* Out     = kbuf + (size_t)bn * Cc * HWc;

  f32x8 acc[4] = {};
  stage_issue(sbuf[0], X, tid);
#pragma unroll 1
  for (int ch = 0; ch < 2; ++ch) {            // K = 64
    stage_wait(tid);
    if (ch + 1 < 2) stage_issue(sbuf[(ch + 1) & 1], X + (ch + 1) * 32 * HWc, tid);
    bf16x16 bf = b_frag_lds(sbuf[ch & 1], PITCH_X, pcol, lane);
#pragma unroll
    for (int mb = 0; mb < 4; ++mb)
      acc[mb] = wmma_bf16(load_a_packed(Apack, ch, mb, lane), bf, acc[mb]);
  }
  store_tile(Out, pblk + wave * 16, acc, benc, lane, false);
}

// ---------------------------------------------------------------------------
// Kernel 3: values = 3x3 conv(dec, W_val) + b_val (implicit GEMM, 9 taps).
// blockIdx.x = row h. 12 double-buffered stages over {3 rows} x {4 K-chunks}.
// ---------------------------------------------------------------------------
__device__ __forceinline__ void stage_issue_v(float* sv, const float* X, int h,
                                              int si, int tid) {
  const int h2 = h + (si >> 2) - 1;
  const int k0 = (si & 3) * 32;
  const bool rowvalid = (unsigned)h2 < (unsigned)Hc;
#if HAVE_TDM
  if (rowvalid) {
    if (tid < 32)  // data at col 1: pad interval=128 DW (6), amount=3 DW (2)
      tdm_stage(lds_addr_of(sv) + 4u, X + k0 * HWc + h2 * Wc, 128u, 32u,
                (unsigned long long)HWc, (1u << 20) | (6u << 22) | (2u << 25));
  } else {
#pragma unroll
    for (int pass = 0; pass < 16; ++pass) {
      const int idx = tid + pass * 256;
      sv[(idx >> 7) * PITCH_V + 1 + (idx & 127)] = 0.f;
    }
  }
#else
#pragma unroll
  for (int pass = 0; pass < 16; ++pass) {
    const int idx = tid + pass * 256;
    const int r = idx >> 7, cw = idx & 127;
    sv[r * PITCH_V + 1 + cw] = rowvalid ? X[(k0 + r) * HWc + h2 * Wc + cw] : 0.f;
  }
#endif
}

__global__ __launch_bounds__(256, 1) void k_values(const float* __restrict__ dec,
                                                   const __bf16* __restrict__ Apack,
                                                   const float* __restrict__ bval,
                                                   float* __restrict__ vbuf) {
  __shared__ float sv[2][32 * PITCH_V];
  const int tid = threadIdx.x, lane = tid & 31, wave = tid >> 5;
  const int b = blockIdx.y;
  const int h = blockIdx.x;
  const int wcol = wave * 16 + (lane & 15);
  const float* X = dec + (size_t)b * DCc * HWc;
  float* Out     = vbuf + (size_t)b * Cc * HWc;

  // Zero halo columns (0, 129) and pad column (130) in both buffers once.
  if (tid < 192) {
    const int buf = tid >= 96;
    const int t   = tid - buf * 96;
    const int r   = t & 31;
    const int cs  = t >> 5;
    sv[buf][r * PITCH_V + (cs == 0 ? 0 : (cs == 1 ? 129 : 130))] = 0.f;
  }

  f32x8 acc[4] = {};
  stage_issue_v(sv[0], X, h, 0, tid);
#pragma unroll 1
  for (int si = 0; si < 12; ++si) {           // si = (row dr)*4 + (K chunk c)
    stage_wait(tid);
    if (si + 1 < 12) stage_issue_v(sv[(si + 1) & 1], X, h, si + 1, tid);
    const int dr = si >> 2, c = si & 3;
    const float* sb = sv[si & 1];
#pragma unroll 1
    for (int dj = 0; dj < 3; ++dj) {
      bf16x16 bf = b_frag_lds(sb, PITCH_V, wcol + dj, lane);
      const int cc = (dr * 3 + dj) * 4 + c;   // packed W_val chunk id
#pragma unroll
      for (int mb = 0; mb < 4; ++mb)
        acc[mb] = wmma_bf16(load_a_packed(Apack, cc, mb, lane), bf, acc[mb]);
    }
  }
  store_tile(Out, h * Wc + wave * 16, acc, bval, lane, false);
}

// ---------------------------------------------------------------------------
// Kernel 4: additive-tanh attention, softmax over 9 taps, weighted gather.
// One thread per pixel, fp32 VALU.
// ---------------------------------------------------------------------------
__global__ __launch_bounds__(256) void k_attn(const float* __restrict__ ctxs,
                                              const float* __restrict__ qbuf,
                                              const float* __restrict__ kbuf,
                                              const float* __restrict__ wagg,
                                              const float* __restrict__ bagg,
                                              float* __restrict__ attbuf) {
  const int p = blockIdx.x * 256 + threadIdx.x;
  const int n = blockIdx.y, b = blockIdx.z;
  const int h = p >> 7, w = p & (Wc - 1);
  const float* q   = qbuf + (size_t)b * Cc * HWc;
  const float* key = kbuf + (size_t)(b * NCc + n) * Cc * HWc;
  const float* ctx = ctxs + (size_t)(b * NCc + n) * Cc * HWc;
  float* att       = attbuf + (size_t)(b * NCc + n) * Cc * HWc;

  int  p2[9];
  bool val[9];
  float sc[9];
  const float bscal = bagg[0];
#pragma unroll
  for (int t = 0; t < 9; ++t) {
    const int h2 = h + t / 3 - 1, w2 = w + t % 3 - 1;
    val[t] = ((unsigned)h2 < (unsigned)Hc) && ((unsigned)w2 < (unsigned)Wc);
    p2[t]  = h2 * Wc + w2;
    sc[t]  = bscal;
  }
#pragma unroll 1
  for (int c = 0; c < Cc; ++c) {
    const float qv = q[c * HWc + p];
    const float wa = wagg[c];
#pragma unroll
    for (int t = 0; t < 9; ++t) {
      const float kv = val[t] ? key[c * HWc + p2[t]] : 0.f;
      sc[t] += tanhf(kv + qv) * wa;
    }
  }
  float mx = sc[0];
#pragma unroll
  for (int t = 1; t < 9; ++t) mx = fmaxf(mx, sc[t]);
  float wgt[9];
  float se = 0.f;
#pragma unroll
  for (int t = 0; t < 9; ++t) { wgt[t] = __expf(sc[t] - mx); se += wgt[t]; }
  const float inv = 1.f / se;
#pragma unroll
  for (int t = 0; t < 9; ++t) wgt[t] *= inv;
#pragma unroll 1
  for (int c = 0; c < Cc; ++c) {
    float s = 0.f;
#pragma unroll
    for (int t = 0; t < 9; ++t) {
      const float cv = val[t] ? ctx[c * HWc + p2[t]] : 0.f;
      s = fmaf(cv, wgt[t], s);
    }
    att[c * HWc + p] = s;
  }
}

// ---------------------------------------------------------------------------
// Kernel 5: out = LeakyReLU(W_attn * cat(values, attended) + b_attn)
// K = 320 = 2 chunks from values + 8 chunks from attended. Double-buffered.
// ---------------------------------------------------------------------------
__global__ __launch_bounds__(256, 1) void k_out(const float* __restrict__ vbuf,
                                                const float* __restrict__ attbuf,
                                                const __bf16* __restrict__ Apack,
                                                const float* __restrict__ battn,
                                                float* __restrict__ out) {
  __shared__ float sbuf[2][32 * PITCH_X];
  const int tid = threadIdx.x, lane = tid & 31, wave = tid >> 5;
  const int b    = blockIdx.y;
  const int pblk = blockIdx.x * 128;
  const int pcol = wave * 16 + (lane & 15);
  const float* V = vbuf   + (size_t)b * Cc * HWc + pblk;
  const float* A = attbuf + (size_t)b * NCc * Cc * HWc + pblk;
  float* O       = out    + (size_t)b * Cc * HWc;

  f32x8 acc[4] = {};
  stage_issue(sbuf[0], V, tid);
#pragma unroll 1
  for (int ch = 0; ch < 10; ++ch) {
    stage_wait(tid);
    if (ch + 1 < 10) {
      const int k1 = (ch + 1) * 32;
      const float* src = (k1 < Cc) ? (V + (size_t)k1 * HWc)
                                   : (A + (size_t)(k1 - Cc) * HWc);
      stage_issue(sbuf[(ch + 1) & 1], src, tid);
    }
    bf16x16 bf = b_frag_lds(sbuf[ch & 1], PITCH_X, pcol, lane);
#pragma unroll
    for (int mb = 0; mb < 4; ++mb)
      acc[mb] = wmma_bf16(load_a_packed(Apack, ch, mb, lane), bf, acc[mb]);
  }
  store_tile(O, pblk + wave * 16, acc, battn, lane, true);
}

// ---------------------------------------------------------------------------
// Host launch
// ---------------------------------------------------------------------------
extern "C" void kernel_launch(void* const* d_in, const int* in_sizes, int n_in,
                              void* d_out, int out_size, void* d_ws, size_t ws_size,
                              hipStream_t stream) {
  const float* contexts = (const float*)d_in[0];   // [B,NC,C,H,W]
  const float* decoded  = (const float*)d_in[1];   // [B,DC,H,W]
  const float* W_enc    = (const float*)d_in[2];   // [C,C]
  const float* b_enc    = (const float*)d_in[3];
  const float* W_dec    = (const float*)d_in[4];   // [C,DC]
  const float* b_dec    = (const float*)d_in[5];
  const float* w_agg    = (const float*)d_in[6];
  const float* b_agg    = (const float*)d_in[7];
  const float* W_attn   = (const float*)d_in[8];   // [C,(NC+1)*C]
  const float* b_attn   = (const float*)d_in[9];
  const float* W_val    = (const float*)d_in[10];  // [C,DC,3,3]
  const float* b_val    = (const float*)d_in[11];
  float* out = (float*)d_out;                      // [B,C,H,W]

  // Workspace: f32 intermediates then bf16 packed weights.
  float* qbuf   = (float*)d_ws;                          // B*C*HW
  float* kbuf   = qbuf + (size_t)Bc * Cc * HWc;          // B*NC*C*HW
  float* vbuf   = kbuf + (size_t)Bc * NCc * Cc * HWc;    // B*C*HW
  float* attbuf = vbuf + (size_t)Bc * Cc * HWc;          // B*NC*C*HW
  __bf16* pk      = (__bf16*)(attbuf + (size_t)Bc * NCc * Cc * HWc);
  __bf16* pk_dec  = pk;                   // 4 chunks * 2048
  __bf16* pk_enc  = pk_dec + 4 * 2048;    // 2 chunks
  __bf16* pk_attn = pk_enc + 2 * 2048;    // 10 chunks
  __bf16* pk_val  = pk_attn + 10 * 2048;  // 36 chunks

  const dim3 blk(256);
  k_pack_plain<<<dim3(4 * 8),  blk, 0, stream>>>(W_dec,  pk_dec,  DCc);
  k_pack_plain<<<dim3(2 * 8),  blk, 0, stream>>>(W_enc,  pk_enc,  Cc);
  k_pack_plain<<<dim3(10 * 8), blk, 0, stream>>>(W_attn, pk_attn, (NCc + 1) * Cc);
  k_pack_wval <<<dim3(36 * 8), blk, 0, stream>>>(W_val,  pk_val);

  k_queries<<<dim3(HWc / 128, Bc),       blk, 0, stream>>>(decoded, pk_dec, b_dec, qbuf);
  k_keys   <<<dim3(HWc / 128, Bc * NCc), blk, 0, stream>>>(contexts, pk_enc, b_enc, kbuf);
  k_values <<<dim3(Hc, Bc),              blk, 0, stream>>>(decoded, pk_val, b_val, vbuf);
  k_attn   <<<dim3(HWc / 256, NCc, Bc),  blk, 0, stream>>>(contexts, qbuf, kbuf, w_agg, b_agg, attbuf);
  k_out    <<<dim3(HWc / 128, Bc),       blk, 0, stream>>>(vbuf, attbuf, pk_attn, b_attn, out);
}
